// GAT_43782896615721
// MI455X (gfx1250) — matrix operand order repaired and checked by reading, not compiled, not run on hardware
//
#include <hip/hip_runtime.h>
#include <hip/hip_bf16.h>

// ---------------------------------------------------------------------------
// GATv2 (3 layers) + per-graph mean readout for MI455X (gfx1250, wave32).
//  - Dense projections: bf16 WMMA GEMM (v_wmma_f32_16x16x32_bf16), fp32 acc.
//    128x64 block tile, 8 waves in 4x2 grid, each wave 32x32 (4 WMMAs/kstep).
//    A staged row-major, B staged column-major in LDS so every fragment is
//    two contiguous ds_load_b128. global_prefetch for the next k-tile.
//  - Edge phase: wave-per-(edge,head), coalesced 64-float head slices,
//    shfl reduction, fp32 atomics for edge-softmax and aggregation.
// ---------------------------------------------------------------------------

typedef __attribute__((ext_vector_type(16))) __bf16 v16bf;
typedef __attribute__((ext_vector_type(8)))  float  v8f;
typedef unsigned short u16;

#define NNODES 50000
#define NEDGES 600000
#define HEADS  4
#define DH     64
#define CH     (HEADS * DH)   /* 256 */
#define NGRAPH 8
#define NEG_SLOPE 0.2f

__device__ __forceinline__ u16 f32_to_bf16(float f) {
  unsigned int u = __float_as_uint(f);
  u += 0x7FFFu + ((u >> 16) & 1u);   // round-to-nearest-even
  return (u16)(u >> 16);
}

// ---------------------------------------------------------------- utilities
__global__ void cvt_f32_bf16_kernel(const float* __restrict__ in,
                                    u16* __restrict__ out, size_t n) {
  size_t i = (size_t)blockIdx.x * blockDim.x + threadIdx.x;
  if (i < n) out[i] = f32_to_bf16(in[i]);
}

__global__ void fill_f32_kernel(float* __restrict__ p, float v, size_t n) {
  size_t i = (size_t)blockIdx.x * blockDim.x + threadIdx.x;
  if (i < n) p[i] = v;
}

// ------------------------------------------------------------- WMMA GEMM
// C[M x 256] = A[M x K](bf16) * B[K x 256](bf16), fp32 accumulate.
// Block: 256 threads = 8 waves. Tile BM=128 x BN=64, BK=32.
// Wave (wm 0..3, wn 0..1) owns a 32x32 sub-tile: 2 A frags x 2 B frags.
#define BM 128
#define BN 64
#define BK 32
#define LDA  40   // ushorts per A row  (32 + 8 pad) = 80 B, 16B-aligned rows
#define LDBC 40   // ushorts per B col  (32 + 8 pad) = 80 B, 16B-aligned cols

__global__ void __launch_bounds__(256)
gemm_bf16_wmma_kernel(const u16* __restrict__ A, const u16* __restrict__ B,
                      float* __restrict__ C, int M, int K) {
  __shared__ __align__(16) u16 As[BM * LDA];    // row-major  [row][k]
  __shared__ __align__(16) u16 Bs[BN * LDBC];   // col-major  [col][k]

  const int t    = threadIdx.x;
  const int lane = t & 31;
  const int wave = t >> 5;
  const int wm   = wave >> 1;          // 0..3 -> 32-row band
  const int wn   = wave & 1;           // 0..1 -> 32-col band
  const int rowBase = blockIdx.x * BM;
  const int colBase = blockIdx.y * BN;

  const int lm    = lane & 15;         // A frag row / B frag col within 16
  const int khalf = lane >> 4;         // K-half select per ISA layout

  v8f c00 = {}, c01 = {}, c10 = {}, c11 = {};

  for (int k0 = 0; k0 < K; k0 += BK) {
    // --- stage A: 128x32; thread -> 16 contiguous bf16 of one row
    {
      int r  = t >> 1;                 // 0..127
      int ch = (t & 1) << 4;           // 0 or 16
      int grow = rowBase + r;
      uint4 q0 = {0, 0, 0, 0}, q1 = {0, 0, 0, 0};
      if (grow < M) {
        const uint4* p = (const uint4*)(A + (size_t)grow * K + k0 + ch);
        q0 = p[0];
        q1 = p[1];
      }
      uint4* d = (uint4*)&As[r * LDA + ch];
      d[0] = q0;
      d[1] = q1;
    }
    // --- stage B: 32x64 row-major global -> column-major LDS scatter
    {
      int r = t >> 3;                  // k row 0..31
      int c = (t & 7) << 3;            // col 0..56
      union { uint4 q; u16 s[8]; } bv;
      bv.q = *(const uint4*)(B + (size_t)(k0 + r) * CH + colBase + c);
#pragma unroll
      for (int i = 0; i < 8; ++i) Bs[(c + i) * LDBC + r] = bv.s[i];
    }
    // --- prefetch next k-tile into cache while we compute this one
    if (k0 + BK < K) {
      int r = t >> 1;
      if (rowBase + r < M)
        __builtin_prefetch(A + (size_t)(rowBase + r) * K + k0 + BK + ((t & 1) << 4), 0, 1);
      __builtin_prefetch(B + (size_t)(k0 + BK + (t >> 3)) * CH + colBase + ((t & 7) << 3), 0, 1);
    }
    __syncthreads();

    // --- fragments: all contiguous 16B-aligned LDS reads (ds_load_b128 x2)
    union { v16bf v; uint4 q[2]; } a0, a1, b0, b1;
    {
      const int r0 = (wm * 32 + lm) * LDA;
      const int r1 = (wm * 32 + 16 + lm) * LDA;
      a0.q[0] = *(const uint4*)&As[r0 + (khalf << 3)];
      a0.q[1] = *(const uint4*)&As[r0 + 16 + (khalf << 3)];
      a1.q[0] = *(const uint4*)&As[r1 + (khalf << 3)];
      a1.q[1] = *(const uint4*)&As[r1 + 16 + (khalf << 3)];
      const int cb0 = (wn * 32 + lm) * LDBC + (khalf << 3);
      const int cb1 = (wn * 32 + 16 + lm) * LDBC + (khalf << 3);
      b0.q[0] = *(const uint4*)&Bs[cb0];
      b0.q[1] = *(const uint4*)&Bs[cb0 + 16];
      b1.q[0] = *(const uint4*)&Bs[cb1];
      b1.q[1] = *(const uint4*)&Bs[cb1 + 16];
    }

    c00 = __builtin_amdgcn_wmma_f32_16x16x32_bf16(false, a0.v, false, b0.v, (short)0, c00, false, false);
    c01 = __builtin_amdgcn_wmma_f32_16x16x32_bf16(false, a0.v, false, b1.v, (short)0, c01, false, false);
    c10 = __builtin_amdgcn_wmma_f32_16x16x32_bf16(false, a1.v, false, b0.v, (short)0, c10, false, false);
    c11 = __builtin_amdgcn_wmma_f32_16x16x32_bf16(false, a1.v, false, b1.v, (short)0, c11, false, false);

    __syncthreads();
  }

  // --- store D. VGPR v -> lanes 0-15: (M=v, N=lane), lanes 16-31: (M=v+8)
  const int madd = khalf << 3;
  const int row0 = rowBase + wm * 32 + madd;
  const int col  = colBase + wn * 32 + lm;
  if (rowBase + BM <= M) {             // block-uniform fast path (common case)
#pragma unroll
    for (int v = 0; v < 8; ++v) {
      C[(size_t)(row0 + v) * CH + col]           = c00[v];
      C[(size_t)(row0 + v) * CH + col + 16]      = c01[v];
      C[(size_t)(row0 + v + 16) * CH + col]      = c10[v];
      C[(size_t)(row0 + v + 16) * CH + col + 16] = c11[v];
    }
  } else {
#pragma unroll
    for (int v = 0; v < 8; ++v) {
      if (row0 + v < M) {
        C[(size_t)(row0 + v) * CH + col]      = c00[v];
        C[(size_t)(row0 + v) * CH + col + 16] = c01[v];
      }
      if (row0 + v + 16 < M) {
        C[(size_t)(row0 + v + 16) * CH + col]      = c10[v];
        C[(size_t)(row0 + v + 16) * CH + col + 16] = c11[v];
      }
    }
  }
}

// ---------------------------------------------------------------- edge phase
__device__ __forceinline__ void atomicMaxF32(float* addr, float val) {
  if (val >= 0.f) atomicMax((int*)addr, __float_as_int(val));
  else            atomicMin((unsigned int*)addr, __float_as_uint(val));
}

// wave per (edge, head): scores[e,h] = a_h . leaky_relu(f[src]+f[dst]),
// plus running max into mbuf[dst,h].
__global__ void __launch_bounds__(256)
edge_scores_kernel(const float* __restrict__ feat, const float* __restrict__ a,
                   const int* __restrict__ src, const int* __restrict__ dst,
                   float* __restrict__ scores, float* __restrict__ mbuf) {
  size_t gtid = (size_t)blockIdx.x * 256 + threadIdx.x;
  size_t w = gtid >> 5;
  int lane = threadIdx.x & 31;
  if (w >= (size_t)NEDGES * HEADS) return;
  int e = (int)(w >> 2);
  int h = (int)(w & 3);
  int s = src[e], d = dst[e];
  const float* fs = feat + (size_t)s * CH + h * DH;
  const float* fd = feat + (size_t)d * CH + h * DH;
  const float* ah = a + h * DH;
  float acc = 0.f;
#pragma unroll
  for (int i = 0; i < 2; ++i) {
    int dd = lane + i * 32;
    float x = fs[dd] + fd[dd];
    x = (x > 0.f) ? x : x * NEG_SLOPE;
    acc += x * ah[dd];
  }
#pragma unroll
  for (int off = 16; off > 0; off >>= 1)
    acc += __shfl_down(acc, off, 32);
  if (lane == 0) {
    scores[(size_t)e * HEADS + h] = acc;
    atomicMaxF32(&mbuf[(size_t)d * HEADS + h], acc);
  }
}

// thread per (edge, head): ex = exp(score - max), denom[dst,h] += ex
__global__ void edge_exp_kernel(const int* __restrict__ dst,
                                const float* __restrict__ mbuf,
                                float* __restrict__ scores,
                                float* __restrict__ denom) {
  size_t i = (size_t)blockIdx.x * blockDim.x + threadIdx.x;
  if (i >= (size_t)NEDGES * HEADS) return;
  int e = (int)(i >> 2);
  int h = (int)(i & 3);
  int d = dst[e];
  float ex = __expf(scores[i] - mbuf[(size_t)d * HEADS + h]);
  scores[i] = ex;
  atomicAdd(&denom[(size_t)d * HEADS + h], ex);
}

// wave per (edge, head): out[dst,h,:] += (ex/denom) * feat[src,h,:]
__global__ void __launch_bounds__(256)
edge_aggregate_kernel(const float* __restrict__ feat,
                      const int* __restrict__ src, const int* __restrict__ dst,
                      const float* __restrict__ ex, const float* __restrict__ denom,
                      float* __restrict__ out) {
  size_t gtid = (size_t)blockIdx.x * 256 + threadIdx.x;
  size_t w = gtid >> 5;
  int lane = threadIdx.x & 31;
  if (w >= (size_t)NEDGES * HEADS) return;
  int e = (int)(w >> 2);
  int h = (int)(w & 3);
  int s = src[e], d = dst[e];
  float alpha = ex[(size_t)e * HEADS + h] / denom[(size_t)d * HEADS + h];
  const float* fs = feat + (size_t)s * CH + h * DH;
  float* od = out + (size_t)d * CH + h * DH;
#pragma unroll
  for (int i = 0; i < 2; ++i) {
    int dd = lane + i * 32;
    atomicAdd(&od[dd], alpha * fs[dd]);
  }
}

// thread per (node, channel): h = elu(aggr [+ residual]); also emit bf16 copy
__global__ void node_finalize_kernel(const float* __restrict__ aggr,
                                     float* __restrict__ hcur,
                                     u16* __restrict__ hbf,
                                     int residual) {
  size_t i = (size_t)blockIdx.x * blockDim.x + threadIdx.x;
  if (i >= (size_t)NNODES * CH) return;
  float v = aggr[i];
  if (residual) v += hcur[i];
  v = (v > 0.f) ? v : (__expf(v) - 1.f);
  hcur[i] = v;
  hbf[i] = f32_to_bf16(v);
}

// ---------------------------------------------------------------- readout
__global__ void __launch_bounds__(256)
graph_sums_kernel(const float* __restrict__ h, const int* __restrict__ gid,
                  float* __restrict__ sums) {
  __shared__ float ls[NGRAPH * CH];   // each thread owns channel c across graphs
  int c = threadIdx.x;                // 0..255
#pragma unroll
  for (int g = 0; g < NGRAPH; ++g) ls[g * CH + c] = 0.f;
  int start = blockIdx.x * 64;
  int end = start + 64;
  if (end > NNODES) end = NNODES;
  for (int i = start; i < end; ++i) {
    int g = gid[i];
    ls[g * CH + c] += h[(size_t)i * CH + c];
  }
#pragma unroll
  for (int g = 0; g < NGRAPH; ++g)
    atomicAdd(&sums[g * CH + c], ls[g * CH + c]);
}

__global__ void graph_counts_kernel(const int* __restrict__ gid,
                                    float* __restrict__ counts) {
  int i = blockIdx.x * blockDim.x + threadIdx.x;
  if (i < NNODES) atomicAdd(&counts[gid[i]], 1.f);
}

__global__ void graph_mean_kernel(const float* __restrict__ sums,
                                  const float* __restrict__ counts,
                                  float* __restrict__ out) {
  int i = blockIdx.x * blockDim.x + threadIdx.x;
  if (i < NGRAPH * CH) {
    float c = counts[i / CH];
    out[i] = sums[i] / fmaxf(c, 1.f);
  }
}

// ---------------------------------------------------------------- launcher
extern "C" void kernel_launch(void* const* d_in, const int* in_sizes, int n_in,
                              void* d_out, int out_size, void* d_ws, size_t ws_size,
                              hipStream_t stream) {
  (void)in_sizes; (void)n_in; (void)out_size; (void)ws_size;
  const float* feat = (const float*)d_in[0];
  const float* W[3] = {(const float*)d_in[1], (const float*)d_in[2], (const float*)d_in[3]};
  const float* a[3] = {(const float*)d_in[4], (const float*)d_in[5], (const float*)d_in[6]};
  const int* src = (const int*)d_in[7];
  const int* dst = (const int*)d_in[8];
  const int* gid = (const int*)d_in[9];

  // bump allocator over workspace
  char* ws = (char*)d_ws;
  size_t off = 0;
  auto take = [&](size_t bytes) -> char* {
    char* p = ws + off;
    off += (bytes + 255) & ~(size_t)255;
    return p;
  };
  float* hcur   = (float*)take((size_t)NNODES * CH * 4);      // fp32 activations
  u16*   hbf    = (u16*)  take((size_t)NNODES * CH * 2);      // bf16 GEMM input
  float* featL  = (float*)take((size_t)NNODES * CH * 4);      // W-projected feats
  float* aggr   = (float*)take((size_t)NNODES * CH * 4);      // scatter target
  float* mbuf   = (float*)take((size_t)NNODES * HEADS * 4);   // softmax max
  float* denom  = (float*)take((size_t)NNODES * HEADS * 4);   // softmax denom
  float* ex     = (float*)take((size_t)NEDGES * HEADS * 4);   // scores -> exp
  u16*   wb     = (u16*)  take((size_t)CH * CH * 2);          // bf16 weights
  float* sums   = (float*)take((size_t)NGRAPH * CH * 4);
  float* counts = (float*)take((size_t)NGRAPH * 4);

  const int TB = 256;
  const size_t nEH  = (size_t)NEDGES * HEADS;                 // 2.4M
  const size_t nNC  = (size_t)NNODES * CH;                    // 12.8M
  const size_t nNH  = (size_t)NNODES * HEADS;
  const int edgeWaveBlocks = (int)((nEH * 32 + TB - 1) / TB); // 300000

  for (int l = 0; l < 3; ++l) {
    const int Kin = (l == 0) ? 128 : CH;
    // weights -> bf16
    size_t wn = (size_t)Kin * CH;
    cvt_f32_bf16_kernel<<<(int)((wn + TB - 1) / TB), TB, 0, stream>>>(W[l], wb, wn);
    if (l == 0) {
      size_t hn = (size_t)NNODES * 128;
      cvt_f32_bf16_kernel<<<(int)((hn + TB - 1) / TB), TB, 0, stream>>>(feat, hbf, hn);
    }
    // projection: featL = h @ W   (bf16 WMMA, fp32 accumulate)
    gemm_bf16_wmma_kernel<<<dim3((NNODES + BM - 1) / BM, CH / BN), TB, 0, stream>>>(
        hbf, wb, featL, NNODES, Kin);

    // per-layer state init
    fill_f32_kernel<<<(int)((nNH + TB - 1) / TB), TB, 0, stream>>>(
        mbuf, -__builtin_huge_valf(), nNH);
    fill_f32_kernel<<<(int)((nNH + TB - 1) / TB), TB, 0, stream>>>(denom, 0.f, nNH);
    fill_f32_kernel<<<(int)((nNC + TB - 1) / TB), TB, 0, stream>>>(aggr, 0.f, nNC);

    // edge softmax + aggregation
    edge_scores_kernel<<<edgeWaveBlocks, TB, 0, stream>>>(featL, a[l], src, dst, ex, mbuf);
    edge_exp_kernel<<<(int)((nEH + TB - 1) / TB), TB, 0, stream>>>(dst, mbuf, ex, denom);
    edge_aggregate_kernel<<<edgeWaveBlocks, TB, 0, stream>>>(featL, src, dst, ex, denom, aggr);

    // residual + elu, emit fp32 + bf16 activations for next layer
    node_finalize_kernel<<<(int)((nNC + TB - 1) / TB), TB, 0, stream>>>(
        aggr, hcur, hbf, l > 0 ? 1 : 0);
  }

  // per-graph mean readout
  fill_f32_kernel<<<(NGRAPH * CH + TB - 1) / TB, TB, 0, stream>>>(sums, 0.f, NGRAPH * CH);
  fill_f32_kernel<<<1, TB, 0, stream>>>(counts, 0.f, NGRAPH);
  graph_counts_kernel<<<(NNODES + TB - 1) / TB, TB, 0, stream>>>(gid, counts);
  graph_sums_kernel<<<(NNODES + 63) / 64, TB, 0, stream>>>(hcur, gid, sums);
  graph_mean_kernel<<<(NGRAPH * CH + TB - 1) / TB, TB, 0, stream>>>(
      sums, counts, (float*)d_out);
}
